// MLPMixerLayer_63986422775878
// MI455X (gfx1250) — matrix operand order repaired and checked by reading, not compiled, not run on hardware
//
#include <hip/hip_runtime.h>
#include <math.h>

// ---------------- CDNA5 (gfx1250) fused MLP-Mixer token-MLP layer ----------
// out = LayerNorm( GELU(X @ W1 + b1) @ W2 + b2 + X ) per token t.
//
// Kernel 1 (prep): W1/W2 f32 -> bf16, transposed + chunk-major into d_ws so
//                  every K-chunk is a contiguous 48KB LDS image.
// Kernel 2 (main): BF16 WMMA (v_wmma_f32_16x16x32_bf16), f32 accumulate.
//                  Weight chunks stream to LDS via GLOBAL_LOAD_ASYNC_TO_LDS in
//                  GVS mode (SGPR base + IOFFSET applies to both LDS & global
//                  sides -> zero per-lane address math), double-buffered so
//                  chunk c+1 DMA overlaps chunk c WMMA. Fused epilogues.

typedef __bf16 bf16;
typedef bf16  v16bf __attribute__((ext_vector_type(16)));
typedef bf16  v8bf  __attribute__((ext_vector_type(8)));
typedef float v8f   __attribute__((ext_vector_type(8)));

#define HDIM 768
#define TTOK 16
#define BROW 1024
#define KC   32             // WMMA K per step (16x16x32 bf16)
#define CHUNKS (HDIM / KC)  // 24
#define MT   32             // rows (B dim) per workgroup (2 row-tiles)
#define THREADS 512         // 16 wave32
#define NT_PER_WAVE 6       // 48 col tiles / 8 col-wave-groups

#define A_STRIDE  776       // bf16 elems per A/H row: 1552 B (16B multiple)
#define BT_STRIDE 32        // bf16 elems per W^T column (contiguous chunk image)

#define WCHUNK_ELEMS (HDIM * KC)                   // 24576 bf16 = 49152 B
#define WCHUNK_BYTES (WCHUNK_ELEMS * 2)
#define WTOK_ELEMS   (CHUNKS * WCHUNK_ELEMS)       // per (layer, token)

#define ABUF_BYTES (MT * A_STRIDE * 2)    // 49664
#define HBUF_BYTES (MT * A_STRIDE * 2)    // 49664
#define BBUF_BYTES WCHUNK_BYTES           // 49152 (x2, double buffered)
#define SMEM_BYTES (ABUF_BYTES + HBUF_BYTES + 2 * BBUF_BYTES)  // 197632 < 320KB

// ============================ prep kernel ==================================
// One block per (layer, token, kchunk): 2*16*24 = 768 blocks.
// dst[((L*16+t)*24+c)*768 + n][kk] = bf16( W[t][c*32+kk][n] )
__global__ __launch_bounds__(256)
void prep_weights_kernel(const float* __restrict__ W1,
                         const float* __restrict__ W2,
                         bf16* __restrict__ ws) {
  const int bid = blockIdx.x;
  const int c   = bid % CHUNKS;
  const int t   = (bid / CHUNKS) % TTOK;
  const int L   = bid / (CHUNKS * TTOK);
  const float* W = (L ? W2 : W1) + (size_t)t * HDIM * HDIM
                                 + (size_t)c * KC * HDIM;
  bf16* dst = ws + (size_t)bid * WCHUNK_ELEMS;

  for (int n = threadIdx.x; n < HDIM; n += 256) {
    v8bf p[4];
#pragma unroll
    for (int q = 0; q < 4; ++q) {
#pragma unroll
      for (int i = 0; i < 8; ++i)
        p[q][i] = (bf16)W[(size_t)(q * 8 + i) * HDIM + n];  // coalesced across lanes
    }
#pragma unroll
    for (int q = 0; q < 4; ++q)
      *(v8bf*)(dst + (size_t)n * KC + q * 8) = p[q];
  }
}

// ============================ main kernel ==================================
// A-fragment (16-bit A 16x32, contiguous K): lane<16 row=lane, K=kb..kb+15, kb=0
//                                         lane>=16 row=lane-16, same with kb=16
__device__ __forceinline__ v16bf load_afrag(const bf16* __restrict__ Asrc,
                                            int row, int kb, int k0) {
  v8bf lo = *(const v8bf*)(Asrc + row * A_STRIDE + k0 + kb);
  v8bf hi = *(const v8bf*)(Asrc + row * A_STRIDE + k0 + kb + 8);
  v16bf a;
#pragma unroll
  for (int i = 0; i < 8; ++i) { a[i] = lo[i]; a[i + 8] = hi[i]; }
  return a;
}

// DMA one 48KB weight chunk to LDS. GVS mode: IOFFSET is added to BOTH the
// LDS destination (VDST VGPR) and the global source (SADDR + VADDR), so one
// operand set serves all 6 transfers. 512 thr * 6 * 16B = 48KB.
__device__ __forceinline__ void async_chunk(unsigned la, unsigned va,
                                            const bf16* __restrict__ sbase) {
  asm volatile(
      "global_load_async_to_lds_b128 %0, %1, %2\n\t"
      "global_load_async_to_lds_b128 %0, %1, %2 offset:8192\n\t"
      "global_load_async_to_lds_b128 %0, %1, %2 offset:16384\n\t"
      "global_load_async_to_lds_b128 %0, %1, %2 offset:24576\n\t"
      "global_load_async_to_lds_b128 %0, %1, %2 offset:32768\n\t"
      "global_load_async_to_lds_b128 %0, %1, %2 offset:40960"
      :: "v"(la), "v"(va), "s"(sbase) : "memory");
}

// 6 WMMA on chunk c from LDS buffer (c&1).
__device__ __forceinline__ void compute_chunk(const bf16* __restrict__ Asrc,
                                              const bf16* __restrict__ Bbuf,
                                              v8f acc[NT_PER_WAVE], int c,
                                              int row, int kb, int bkb,
                                              int rt, int wc) {
  const bf16* Bc = Bbuf + ((c & 1) ? WCHUNK_ELEMS : 0);
  v16bf a = load_afrag(Asrc + rt * A_STRIDE, row, kb, c * KC);
#pragma unroll
  for (int j = 0; j < NT_PER_WAVE; ++j) {
    int ccol = (wc * NT_PER_WAVE + j) * 16 + row;
    v16bf b = *(const v16bf*)(Bc + ccol * BT_STRIDE + bkb);
    acc[j] = __builtin_amdgcn_wmma_f32_16x16x32_bf16(
        false, a, false, b, (short)0, acc[j], false, false);
  }
}

// K-blocked, double-buffered GEMM stage.
// Pipeline per chunk c: wait(c) -> barrier -> issue(c+1) -> WMMA on c.
// B-fragments: 32x16, col = lane%16, lanes 16-31 hold K=16..31 (contiguous 32B).
__device__ __forceinline__ void gemm_stage(const bf16* __restrict__ Asrc,
                                           const bf16* __restrict__ Wp,
                                           const bf16* __restrict__ Bbuf,
                                           unsigned bbuf_lds,
                                           v8f acc[NT_PER_WAVE],
                                           int tid, int lane, int wave) {
  const int row = lane & 15;
  const int hi  = lane >> 4;
  const int kb  = hi * 16;                   // A-frag K base
  const int bkb = hi * 16;                   // B-frag K base
  const int rt  = (wave >> 3) * 16;          // row-tile base (0 or 16)
  const int wc  = wave & 7;                  // column wave-group

  const unsigned va  = (unsigned)(tid * 16);        // per-lane offset (both sides)
  const unsigned la0 = bbuf_lds + va;               // LDS dest, buffer 0
  const unsigned la1 = la0 + (unsigned)BBUF_BYTES;  // LDS dest, buffer 1

  async_chunk(la0, va, Wp);
  for (int c = 0; c < CHUNKS - 1; ++c) {
    // Only chunk c's 6 per-thread asyncs are outstanding here (in-order done).
    asm volatile("s_wait_asynccnt 0x0" ::: "memory");
    __syncthreads();   // chunk c visible everywhere; all waves done computing c-1
    async_chunk(((c + 1) & 1) ? la1 : la0, va,
                Wp + (size_t)(c + 1) * WCHUNK_ELEMS);  // WAR-safe: last read at c-1
    compute_chunk(Asrc, Bbuf, acc, c, row, kb, bkb, rt, wc);
  }
  asm volatile("s_wait_asynccnt 0x0" ::: "memory");
  __syncthreads();
  compute_chunk(Asrc, Bbuf, acc, CHUNKS - 1, row, kb, bkb, rt, wc);
  __syncthreads();   // protect LDS (and aliases) after last compute
}

__global__ __launch_bounds__(THREADS)
void mixer_token_mlp_ln_kernel(const float* __restrict__ Xg,
                               const bf16* __restrict__ Wbf,   // prepped weights
                               const float* __restrict__ b1,
                               const float* __restrict__ b2,
                               const float* __restrict__ gamma,
                               const float* __restrict__ beta,
                               float* __restrict__ out) {
  extern __shared__ __align__(16) unsigned char smem[];
  bf16*  Abuf = (bf16*)smem;                             // X tile, bf16
  bf16*  Hbuf = (bf16*)(smem + ABUF_BYTES);              // gelu(h) tile, bf16
  bf16*  Bbuf = (bf16*)(smem + ABUF_BYTES + HBUF_BYTES); // weight chunks (x2)
  float* Xbuf = (float*)(smem + ABUF_BYTES);             // pre-LN rows (aliases H+B0)
  // Generic LDS addresses: low 32 bits are the LDS byte offset (ISA aperture rule)
  const unsigned bbuf_lds = (unsigned)(uintptr_t)(void*)Bbuf;

  const int tid  = threadIdx.x;
  const int lane = tid & 31;
  const int wave = tid >> 5;
  const int t    = blockIdx.x / (BROW / MT);  // 32 consecutive blocks share weights
  const int m0   = (blockIdx.x % (BROW / MT)) * MT;

  const bf16* W1p = Wbf + (size_t)t * WTOK_ELEMS;
  const bf16* W2p = Wbf + (size_t)(TTOK + t) * WTOK_ELEMS;
  const float* b1t = b1 + t * HDIM;
  const float* b2t = b2 + t * HDIM;

  const int row = lane & 15;
  const int hi  = lane >> 4;
  const int rt  = (wave >> 3) * 16;
  const int wc  = wave & 7;

  // ---- Stage 0: X rows -> Abuf (bf16) ----
  for (int i = tid; i < MT * HDIM; i += THREADS) {
    int r = i / HDIM, ccl = i % HDIM;
    Abuf[r * A_STRIDE + ccl] = (bf16)Xg[((size_t)(m0 + r) * TTOK + t) * HDIM + ccl];
  }

  v8f acc[NT_PER_WAVE];
  const v8f vzero = {0.f, 0.f, 0.f, 0.f, 0.f, 0.f, 0.f, 0.f};
#pragma unroll
  for (int j = 0; j < NT_PER_WAVE; ++j) acc[j] = vzero;

  // ---- Stage 1: h = X @ W1 ----  (first in-loop barrier covers Abuf fill)
  gemm_stage(Abuf, W1p, Bbuf, bbuf_lds, acc, tid, lane, wave);

  // ---- Epilogue 1: +b1, exact GELU, -> Hbuf bf16 ----
  // acc layout: VGPR v, lanes 0-15 -> M=v, N=lane; lanes 16-31 -> M=v+8, N=lane-16
#pragma unroll
  for (int j = 0; j < NT_PER_WAVE; ++j) {
    int ccol = (wc * NT_PER_WAVE + j) * 16 + row;
    float bias = b1t[ccol];
#pragma unroll
    for (int v = 0; v < 8; ++v) {
      int r = rt + v + 8 * hi;
      float x = acc[j][v] + bias;
      float g = 0.5f * x * (1.0f + erff(x * 0.7071067811865476f));
      Hbuf[r * A_STRIDE + ccol] = (bf16)g;
    }
    acc[j] = vzero;
  }

  // ---- Stage 2: mlp = gelu(h) @ W2 ----  (in-loop barrier orders Hbuf writes)
  gemm_stage(Hbuf, W2p, Bbuf, bbuf_lds, acc, tid, lane, wave);

  // ---- Epilogue 2: +b2 + residual (f32 reload), stash pre-LN rows ----
#pragma unroll
  for (int j = 0; j < NT_PER_WAVE; ++j) {
    int ccol = (wc * NT_PER_WAVE + j) * 16 + row;
    float bias = b2t[ccol];
#pragma unroll
    for (int v = 0; v < 8; ++v) {
      int r = rt + v + 8 * hi;
      float x = acc[j][v] + bias + Xg[((size_t)(m0 + r) * TTOK + t) * HDIM + ccol];
      Xbuf[r * HDIM + ccol] = x;
    }
  }
  __syncthreads();

  // ---- LayerNorm: each wave normalizes 2 rows (wave32 butterfly reduce) ----
#pragma unroll
  for (int rr = 0; rr < 2; ++rr) {
    int r = wave * 2 + rr;
    float s = 0.f, s2 = 0.f;
    for (int ccl = lane; ccl < HDIM; ccl += 32) {
      float v = Xbuf[r * HDIM + ccl];
      s += v; s2 += v * v;
    }
#pragma unroll
    for (int off = 16; off > 0; off >>= 1) {
      s  += __shfl_xor(s,  off, 32);
      s2 += __shfl_xor(s2, off, 32);
    }
    const float inv_h = 1.0f / (float)HDIM;
    float mean = s * inv_h;
    float var  = s2 * inv_h - mean * mean;
    float rstd = rsqrtf(var + 1e-3f);
    for (int ccl = lane; ccl < HDIM; ccl += 32) {
      float v = (Xbuf[r * HDIM + ccl] - mean) * rstd;
      out[((size_t)(m0 + r) * TTOK + t) * HDIM + ccl] = v * gamma[ccl] + beta[ccl];
    }
  }
}

extern "C" void kernel_launch(void* const* d_in, const int* in_sizes, int n_in,
                              void* d_out, int out_size, void* d_ws, size_t ws_size,
                              hipStream_t stream) {
  const float* X     = (const float*)d_in[0];
  const float* W1    = (const float*)d_in[1];
  const float* b1    = (const float*)d_in[2];
  const float* W2    = (const float*)d_in[3];
  const float* b2    = (const float*)d_in[4];
  const float* gamma = (const float*)d_in[5];
  const float* beta  = (const float*)d_in[6];
  float* out = (float*)d_out;
  bf16* wbf  = (bf16*)d_ws;   // needs 37.75 MB scratch

  prep_weights_kernel<<<dim3(2 * TTOK * CHUNKS), dim3(256), 0, stream>>>(W1, W2, wbf);

  dim3 grid(TTOK * (BROW / MT));   // 512 workgroups: t major, row-block minor
  dim3 block(THREADS);
  mixer_token_mlp_ln_kernel<<<grid, block, SMEM_BYTES, stream>>>(
      X, wbf, b1, b2, gamma, beta, out);
}
